// GNN_50792283242911
// MI455X (gfx1250) — matrix operand order repaired and checked by reading, not compiled, not run on hardware
//
#include <hip/hip_runtime.h>
#include <hip/hip_bf16.h>

#define N_NODES 100000
#define N_EDGES 400000
#define NODE_DIM 64
#define EDGE_DIM 16
#define HID 128
#define K_EDGE 272      // 2*HID + EDGE_DIM
#define K_EDGE_PAD 288  // padded to multiple of 32
#define K_NODE 256      // 2*HID

typedef __attribute__((ext_vector_type(16))) __bf16 v16bf;
typedef __attribute__((ext_vector_type(8)))  float  v8f;
typedef __attribute__((ext_vector_type(4)))  int    v4i;

// ---------- bf16 helpers (native converts -> v_cvt_pk_bf16_f32) ----------
__device__ __forceinline__ __bf16 f2bf(float f) { return (__bf16)f; }
__device__ __forceinline__ unsigned int pack2bf(float a, float b) {
  union { __bf16 h[2]; unsigned int u; } v;
  v.h[0] = (__bf16)a; v.h[1] = (__bf16)b;
  return v.u;
}

// ---------- async global->LDS staging (CDNA5 ASYNCcnt path) ----------
#if defined(__has_builtin)
#if __has_builtin(__builtin_amdgcn_global_load_async_to_lds_b128) && \
    __has_builtin(__builtin_amdgcn_s_wait_asynccnt)
#define HAVE_ASYNC_LDS 1
#endif
#endif

__device__ __forceinline__ void async_ld_b128(void* lds, const void* gsrc) {
#ifdef HAVE_ASYNC_LDS
  __builtin_amdgcn_global_load_async_to_lds_b128(
      (__attribute__((address_space(1))) v4i*)(gsrc),
      (__attribute__((address_space(3))) v4i*)(lds), 0, 0);
#else
  *reinterpret_cast<uint4*>(lds) = *reinterpret_cast<const uint4*>(gsrc);
#endif
}
__device__ __forceinline__ void async_wait() {
#ifdef HAVE_ASYNC_LDS
  __builtin_amdgcn_s_wait_asynccnt(0);
#endif
}

// ---------- WMMA fragment loaders ----------
// A: 16(M)x32(K) bf16 tile, row-major in LDS with pitch (halves).
__device__ __forceinline__ v16bf load_a(const __bf16* base, int pitch, int kbase, int lane) {
  int row = lane & 15, hi = (lane >> 4) & 1;
  const __bf16* p = base + row * pitch + kbase + hi * 8;
  v16bf a;
  uint4* au = reinterpret_cast<uint4*>(&a);
  au[0] = *reinterpret_cast<const uint4*>(p);
  au[1] = *reinterpret_cast<const uint4*>(p + 16);
  return a;
}
// B: 32(K)x16(N) bf16 tile from a column-major [Ncols][Kpad] weight image.
__device__ __forceinline__ v16bf load_b(const __bf16* baseT, int pitch, int kbase, int lane) {
  int col = lane & 15, hi = (lane >> 4) & 1;
  const __bf16* p = baseT + col * pitch + kbase + hi * 8;
  v16bf b;
  uint4* bu = reinterpret_cast<uint4*>(&b);
  bu[0] = *reinterpret_cast<const uint4*>(p);
  bu[1] = *reinterpret_cast<const uint4*>(p + 16);
  return b;
}
#define WMMA_BF16(acc, a, b) \
  __builtin_amdgcn_wmma_f32_16x16x32_bf16(false, (a), false, (b), (short)0, (acc), false, false)

__device__ __forceinline__ float silu(float x) { return x / (1.f + __expf(-x)); }

// ---------- utility kernels ----------
__global__ void cvt_bf16_kernel(const float* __restrict__ src, __bf16* __restrict__ dst, int n) {
  for (int i = blockIdx.x * blockDim.x + threadIdx.x; i < n; i += gridDim.x * blockDim.x)
    dst[i] = f2bf(src[i]);
}
// src [K][Ncols] row-major fp32 -> dst [Ncols][Kpad] bf16 (zero-padded K)
__global__ void cvt_transpose_kernel(const float* __restrict__ src, __bf16* __restrict__ dst,
                                     int K, int Kpad, int Ncols) {
  int total = Ncols * Kpad;
  for (int i = blockIdx.x * blockDim.x + threadIdx.x; i < total; i += gridDim.x * blockDim.x) {
    int n = i / Kpad, k = i % Kpad;
    dst[i] = (k < K) ? f2bf(src[(size_t)k * Ncols + n]) : f2bf(0.f);
  }
}
__global__ void zero_kernel(float* __restrict__ p, size_t n) {
  for (size_t i = blockIdx.x * (size_t)blockDim.x + threadIdx.x; i < n;
       i += (size_t)gridDim.x * blockDim.x)
    p[i] = 0.f;
}

// ---------- h = nodes @ W_emb + b_emb ----------
__global__ __launch_bounds__(256) void embed_kernel(
    const float* __restrict__ nodes, const __bf16* __restrict__ WembT,
    const float* __restrict__ bemb, float* __restrict__ h, __bf16* __restrict__ h_bf,
    int nNodes) {
  __shared__ __align__(16) __bf16 xs[16 * NODE_DIM];
  __shared__ float bs[HID];
  int tid = threadIdx.x, lane = tid & 31, wave = tid >> 5;
  if (tid < HID) bs[tid] = bemb[tid];
  __syncthreads();
  int nTiles = nNodes >> 4;
  for (int t = blockIdx.x; t < nTiles; t += gridDim.x) {
    {  // stage & convert one 16x64 fp32 tile -> bf16 LDS
      int e = tid >> 4, s = tid & 15;
      float4 n4 = *reinterpret_cast<const float4*>(nodes + (size_t)(t * 16 + e) * NODE_DIM + s * 4);
      uint2 p;
      p.x = pack2bf(n4.x, n4.y);
      p.y = pack2bf(n4.z, n4.w);
      *reinterpret_cast<uint2*>(xs + e * NODE_DIM + s * 4) = p;
    }
    __syncthreads();
    v8f acc = {};
    const __bf16* bT = WembT + (size_t)(wave * 16) * NODE_DIM;
#pragma unroll
    for (int kc = 0; kc < NODE_DIM / 32; ++kc) {
      v16bf a = load_a(xs, NODE_DIM, kc * 32, lane);
      v16bf b = load_b(bT, NODE_DIM, kc * 32, lane);
      acc = WMMA_BF16(acc, a, b);
    }
    int col = wave * 16 + (lane & 15);
    int rb = (lane >> 4) * 8;
#pragma unroll
    for (int i = 0; i < 8; ++i) {
      int r = t * 16 + rb + i;
      float v = acc[i] + bs[col];
      h[(size_t)r * HID + col] = v;
      h_bf[(size_t)r * HID + col] = f2bf(v);
    }
    __syncthreads();
  }
}

// ---------- fused edge MLP + gate + scatter-add ----------
__global__ __launch_bounds__(256) void edge_kernel(
    const __bf16* __restrict__ h_bf, const __bf16* __restrict__ ef_bf,
    const int* __restrict__ eidx,
    const __bf16* __restrict__ We1T, const float* __restrict__ be1,
    const __bf16* __restrict__ We2T, const float* __restrict__ be2,
    const float* __restrict__ winf_g, const float* __restrict__ binf_g,
    float* __restrict__ agg, int nEdges) {
  __shared__ __align__(16) __bf16 xs[16 * K_EDGE_PAD];  // edge inputs (padded K)
  __shared__ __align__(16) __bf16 m1[16 * HID];         // layer-1 activations (bf16)
  __shared__ __align__(16) float m2f[16 * HID];         // layer-2 activations (fp32)
  __shared__ float gate[16];
  __shared__ int sidx[16], didx[16];
  __shared__ float be1s[HID], be2s[HID];
  __shared__ __align__(16) float winfs[HID];

  int tid = threadIdx.x, lane = tid & 31, wave = tid >> 5;
  if (tid < HID) { be1s[tid] = be1[tid]; be2s[tid] = be2[tid]; winfs[tid] = winf_g[tid]; }
  float binfv = binf_g[0];
  // zero the K padding region once (never touched by async gathers)
  if (tid < 32) {
    int e = tid >> 1, s = tid & 1;
    uint4 z = {0u, 0u, 0u, 0u};
    *reinterpret_cast<uint4*>(xs + e * K_EDGE_PAD + K_EDGE + s * 8) = z;
  }
  __syncthreads();

  int nTiles = nEdges >> 4;
  for (int t = blockIdx.x; t < nTiles; t += gridDim.x) {
    int tn = t + gridDim.x;
    if (tn < nTiles) __builtin_prefetch(&eidx[tn * 16], 0, 0);
    if (tid < 16) {
      sidx[tid] = eidx[t * 16 + tid];
      didx[tid] = eidx[nEdges + t * 16 + tid];
    }
    __syncthreads();
    {  // async gather x = [h_bf[s], h_bf[d], ef] into LDS (2+ B128 per thread)
      int e = tid >> 4, s = tid & 15;
      async_ld_b128(xs + e * K_EDGE_PAD + s * 8, h_bf + (size_t)sidx[e] * HID + s * 8);
      async_ld_b128(xs + e * K_EDGE_PAD + HID + s * 8, h_bf + (size_t)didx[e] * HID + s * 8);
      if (tid < 32) {
        int e2 = tid >> 1, s2 = tid & 1;
        async_ld_b128(xs + e2 * K_EDGE_PAD + 2 * HID + s2 * 8,
                      ef_bf + (size_t)(t * 16 + e2) * EDGE_DIM + s2 * 8);
      }
    }
    async_wait();
    __syncthreads();
    // layer 1: [16 x 272] @ [272 x 128], SiLU
    {
      v8f acc = {};
      const __bf16* bT = We1T + (size_t)(wave * 16) * K_EDGE_PAD;
#pragma unroll
      for (int kc = 0; kc < K_EDGE_PAD / 32; ++kc) {
        v16bf a = load_a(xs, K_EDGE_PAD, kc * 32, lane);
        v16bf b = load_b(bT, K_EDGE_PAD, kc * 32, lane);
        acc = WMMA_BF16(acc, a, b);
      }
      int col = wave * 16 + (lane & 15);
      int rb = (lane >> 4) * 8;
#pragma unroll
      for (int i = 0; i < 8; ++i)
        m1[(rb + i) * HID + col] = f2bf(silu(acc[i] + be1s[col]));
    }
    __syncthreads();
    // layer 2: [16 x 128] @ [128 x 128], SiLU
    {
      v8f acc = {};
      const __bf16* bT = We2T + (size_t)(wave * 16) * HID;
#pragma unroll
      for (int kc = 0; kc < HID / 32; ++kc) {
        v16bf a = load_a(m1, HID, kc * 32, lane);
        v16bf b = load_b(bT, HID, kc * 32, lane);
        acc = WMMA_BF16(acc, a, b);
      }
      int col = wave * 16 + (lane & 15);
      int rb = (lane >> 4) * 8;
#pragma unroll
      for (int i = 0; i < 8; ++i)
        m2f[(rb + i) * HID + col] = silu(acc[i] + be2s[col]);
    }
    __syncthreads();
    // gate = sigmoid(m2 . Winf + binf) (vectorized per edge)
    if (tid < 16) {
      float g = binfv;
      const float4* m4 = reinterpret_cast<const float4*>(m2f + tid * HID);
      const float4* w4 = reinterpret_cast<const float4*>(winfs);
#pragma unroll 4
      for (int k = 0; k < HID / 4; ++k) {
        float4 m = m4[k], w = w4[k];
        g += m.x * w.x + m.y * w.y + m.z * w.z + m.w * w.w;
      }
      gate[tid] = 1.f / (1.f + __expf(-g));
    }
    __syncthreads();
    // scatter-add into agg[start] (L2-resident atomics)
    for (int i = tid; i < 16 * HID; i += 256) {
      int e = i >> 7, k = i & 127;
      atomicAdd(&agg[(size_t)sidx[e] * HID + k], m2f[i] * gate[e]);
    }
    __syncthreads();
  }
}

// ---------- fused node MLP + residual ----------
__global__ __launch_bounds__(256) void node_kernel(
    float* __restrict__ h, __bf16* __restrict__ h_bf, const float* __restrict__ agg,
    const __bf16* __restrict__ Wn1T, const float* __restrict__ bn1,
    const __bf16* __restrict__ Wn2T, const float* __restrict__ bn2, int nNodes) {
  __shared__ __align__(16) __bf16 xs[16 * K_NODE];
  __shared__ __align__(16) __bf16 t1[16 * HID];
  __shared__ float bn1s[HID], bn2s[HID];
  int tid = threadIdx.x, lane = tid & 31, wave = tid >> 5;
  if (tid < HID) { bn1s[tid] = bn1[tid]; bn2s[tid] = bn2[tid]; }
  __syncthreads();
  int nTiles = nNodes >> 4;
  for (int t = blockIdx.x; t < nTiles; t += gridDim.x) {
    {  // stage xn = [h_bf (async), bf16(agg) (convert)]
      int e = tid >> 4, s = tid & 15;
      int row = t * 16 + e;
      async_ld_b128(xs + e * K_NODE + s * 8, h_bf + (size_t)row * HID + s * 8);
      const float4* ag = reinterpret_cast<const float4*>(agg + (size_t)row * HID + s * 8);
      float4 a0 = ag[0], a1 = ag[1];
      uint4 packed;
      packed.x = pack2bf(a0.x, a0.y);
      packed.y = pack2bf(a0.z, a0.w);
      packed.z = pack2bf(a1.x, a1.y);
      packed.w = pack2bf(a1.z, a1.w);
      *reinterpret_cast<uint4*>(xs + e * K_NODE + HID + s * 8) = packed;
    }
    async_wait();
    __syncthreads();
    // layer 1: [16 x 256] @ [256 x 128], SiLU
    {
      v8f acc = {};
      const __bf16* bT = Wn1T + (size_t)(wave * 16) * K_NODE;
#pragma unroll
      for (int kc = 0; kc < K_NODE / 32; ++kc) {
        v16bf a = load_a(xs, K_NODE, kc * 32, lane);
        v16bf b = load_b(bT, K_NODE, kc * 32, lane);
        acc = WMMA_BF16(acc, a, b);
      }
      int col = wave * 16 + (lane & 15);
      int rb = (lane >> 4) * 8;
#pragma unroll
      for (int i = 0; i < 8; ++i)
        t1[(rb + i) * HID + col] = f2bf(silu(acc[i] + bn1s[col]));
    }
    __syncthreads();
    // layer 2 + residual: h += t1 @ Wn2 + bn2
    {
      v8f acc = {};
      const __bf16* bT = Wn2T + (size_t)(wave * 16) * HID;
#pragma unroll
      for (int kc = 0; kc < HID / 32; ++kc) {
        v16bf a = load_a(t1, HID, kc * 32, lane);
        v16bf b = load_b(bT, HID, kc * 32, lane);
        acc = WMMA_BF16(acc, a, b);
      }
      int col = wave * 16 + (lane & 15);
      int rb = (lane >> 4) * 8;
#pragma unroll
      for (int i = 0; i < 8; ++i) {
        int r = t * 16 + rb + i;
        float hn = h[(size_t)r * HID + col] + acc[i] + bn2s[col];
        h[(size_t)r * HID + col] = hn;
        h_bf[(size_t)r * HID + col] = f2bf(hn);
      }
    }
    __syncthreads();
  }
}

// ---------- pooling ----------
__global__ void pool_kernel(const float* __restrict__ h, float* __restrict__ pooled, int n) {
  int g = blockIdx.x * blockDim.x + threadIdx.x;
  int col = g & 127;
  int rstride = (gridDim.x * blockDim.x) >> 7;
  float acc = 0.f;
  for (int r = g >> 7; r < n; r += rstride) acc += h[(size_t)r * HID + col];
  atomicAdd(&pooled[col], acc);
}
__global__ void out_kernel(const float* __restrict__ pooled, const float* __restrict__ Wout,
                           const float* __restrict__ bout, float* __restrict__ out, int n) {
  __shared__ float red[HID];
  int t = threadIdx.x;
  red[t] = (pooled[t] / (float)n) * Wout[t];
  __syncthreads();
  for (int s = 64; s > 0; s >>= 1) {
    if (t < s) red[t] += red[t + s];
    __syncthreads();
  }
  if (t == 0) out[0] = red[0] + bout[0];
}

extern "C" void kernel_launch(void* const* d_in, const int* in_sizes, int n_in,
                              void* d_out, int out_size, void* d_ws, size_t ws_size,
                              hipStream_t stream) {
  const float* nodes = (const float*)d_in[0];
  const int*   eidx  = (const int*)d_in[1];
  const float* ef    = (const float*)d_in[2];
  const float* W_emb = (const float*)d_in[3];
  const float* b_emb = (const float*)d_in[4];
  const float* We1   = (const float*)d_in[5];
  const float* be1   = (const float*)d_in[6];
  const float* We2   = (const float*)d_in[7];
  const float* be2   = (const float*)d_in[8];
  const float* Wn1   = (const float*)d_in[9];
  const float* bn1   = (const float*)d_in[10];
  const float* Wn2   = (const float*)d_in[11];
  const float* bn2   = (const float*)d_in[12];
  const float* Winf  = (const float*)d_in[13];
  const float* binf  = (const float*)d_in[14];
  const float* Wout  = (const float*)d_in[15];
  const float* bout  = (const float*)d_in[16];
  float* out = (float*)d_out;

  char* ws = (char*)d_ws;
  size_t off = 0;
  auto carve = [&](size_t bytes) -> char* {
    char* p = ws + off;
    off = (off + bytes + 255) & ~(size_t)255;
    return p;
  };
  float*  h      = (float*)carve((size_t)N_NODES * HID * 4);
  __bf16* h_bf   = (__bf16*)carve((size_t)N_NODES * HID * 2);
  float*  agg    = (float*)carve((size_t)N_NODES * HID * 4);
  __bf16* ef_bf  = (__bf16*)carve((size_t)N_EDGES * EDGE_DIM * 2);
  __bf16* WembT  = (__bf16*)carve((size_t)HID * NODE_DIM * 2);
  __bf16* We1T   = (__bf16*)carve((size_t)2 * HID * K_EDGE_PAD * 2);
  __bf16* We2T   = (__bf16*)carve((size_t)2 * HID * HID * 2);
  __bf16* Wn1T   = (__bf16*)carve((size_t)2 * HID * K_NODE * 2);
  __bf16* Wn2T   = (__bf16*)carve((size_t)2 * HID * HID * 2);
  float*  pooled = (float*)carve(HID * 4);

  // weight prep: bf16, column-major, K padded
  cvt_transpose_kernel<<<32, 256, 0, stream>>>(W_emb, WembT, NODE_DIM, NODE_DIM, HID);
  for (int blk = 0; blk < 2; ++blk) {
    cvt_transpose_kernel<<<160, 256, 0, stream>>>(We1 + (size_t)blk * K_EDGE * HID,
                                                  We1T + (size_t)blk * HID * K_EDGE_PAD,
                                                  K_EDGE, K_EDGE_PAD, HID);
    cvt_transpose_kernel<<<64, 256, 0, stream>>>(We2 + (size_t)blk * HID * HID,
                                                 We2T + (size_t)blk * HID * HID, HID, HID, HID);
    cvt_transpose_kernel<<<128, 256, 0, stream>>>(Wn1 + (size_t)blk * K_NODE * HID,
                                                  Wn1T + (size_t)blk * HID * K_NODE,
                                                  K_NODE, K_NODE, HID);
    cvt_transpose_kernel<<<64, 256, 0, stream>>>(Wn2 + (size_t)blk * HID * HID,
                                                 Wn2T + (size_t)blk * HID * HID, HID, HID, HID);
  }
  cvt_bf16_kernel<<<2048, 256, 0, stream>>>(ef, ef_bf, N_EDGES * EDGE_DIM);

  embed_kernel<<<1536, 256, 0, stream>>>(nodes, WembT, b_emb, h, h_bf, N_NODES);

  for (int blk = 0; blk < 2; ++blk) {
    zero_kernel<<<2048, 256, 0, stream>>>(agg, (size_t)N_NODES * HID);
    edge_kernel<<<2048, 256, 0, stream>>>(
        h_bf, ef_bf, eidx,
        We1T + (size_t)blk * HID * K_EDGE_PAD, be1 + blk * HID,
        We2T + (size_t)blk * HID * HID, be2 + blk * HID,
        Winf + blk * HID, binf + blk, agg, N_EDGES);
    node_kernel<<<1024, 256, 0, stream>>>(
        h, h_bf, agg,
        Wn1T + (size_t)blk * HID * K_NODE, bn1 + blk * HID,
        Wn2T + (size_t)blk * HID * HID, bn2 + blk * HID, N_NODES);
  }

  zero_kernel<<<1, 128, 0, stream>>>(pooled, HID);
  pool_kernel<<<512, 256, 0, stream>>>(h, pooled, N_NODES);
  out_kernel<<<1, 128, 0, stream>>>(pooled, Wout, bout, out, N_NODES);
}
